// Class_Cross_Attention_V1_17746804867800
// MI455X (gfx1250) — compile-verified, matrix operand order checked by reading, not run on hardware
//
#include <hip/hip_runtime.h>
#include <hip/hip_bf16.h>

// ---------------------------------------------------------------------------
// Class_Cross_Attention_V1 for MI455X (gfx1250, wave32, WMMA).
// B=4, C=512, H=8, hd=64, QL=225, N=196.
// - pointwise conv + mean_n commute -> 900x512x512 GEMM on reduced tensor
// - hs is rank-1 per (b,c) -> fused dwconv/bn/relu/mean, no 361MB intermediate
// - GEMMs: 128x64 block tile, 32x32 register tile per wave -> 4x
//   v_wmma_f32_16x16x32_f16 per wave per K-step, f16 operands staged in LDS
//   via branchless clamped loads + v_cvt_pk_rtz_f16_f32 + ds_store_b32.
// ---------------------------------------------------------------------------

typedef _Float16 v16h __attribute__((ext_vector_type(16)));
typedef _Float16 v8h  __attribute__((ext_vector_type(8)));
typedef __fp16   hh2  __attribute__((ext_vector_type(2)));   // cvt_pkrtz result type
typedef float    v8f  __attribute__((ext_vector_type(8)));

#define CB 4
#define CC 512
#define CH 8
#define CHD 64
#define CQL 225
#define CN 196
#define CSEQ 421            // VS + QL

#define TILE_M 128
#define TILE_N 64
#define TILE_K 32
#define LDSW 40             // halves per LDS row (80 bytes, multiple of 16)

// flags: bit0 = relu on accumulator, bit2 = W is stored [K][N] (transposed)
__global__ __launch_bounds__(256) void gemm_wmma_kernel(
    const float* __restrict__ X, const float* __restrict__ W,
    const float* __restrict__ bias, const float* __restrict__ resid,
    float* __restrict__ Y,
    int M, int Nout, int K,
    int ldx, int ldw, int ldy, int ldr,
    long long sx_o, long long sx_i,
    long long sw_o, long long sw_i,
    long long sy_o, long long sy_i,
    long long sr_o, long long sr_i,
    int innerCount, float alpha, int flags)
{
    __shared__ __attribute__((aligned(16))) _Float16 Xs[TILE_M * LDSW];
    __shared__ __attribute__((aligned(16))) _Float16 Ws[TILE_N * LDSW];

    const int tid = threadIdx.x;
    const int bz  = blockIdx.z;
    const int bo  = bz / innerCount;
    const int bi  = bz % innerCount;

    const float* Xb = X + bo * sx_o + bi * sx_i;
    const float* Wb = W + bo * sw_o + bi * sw_i;
    float*       Yb = Y + bo * sy_o + bi * sy_i;
    const float* Rb = resid ? (resid + bo * sr_o + bi * sr_i) : (const float*)0;

    const int m0 = blockIdx.x * TILE_M;
    const int n0 = blockIdx.y * TILE_N;
    const bool wtrans = (flags & 4) != 0;
    const bool mfull  = (m0 + TILE_M) <= M;
    const bool nfull  = (n0 + TILE_N) <= Nout;

    const int wave  = tid >> 5;
    const int lane  = tid & 31;
    const int mpair = wave & 3;       // -> msubs {2*mpair, 2*mpair+1} of 8
    const int npair = wave >> 2;      // -> nsubs {2*npair, 2*npair+1} of 4
    const int lrow  = lane & 15;
    const int hsel  = lane >> 4;      // K-half selector per ISA A/B layout

    v8f acc[2][2];
    #pragma unroll
    for (int mi = 0; mi < 2; ++mi)
        #pragma unroll
        for (int ni = 0; ni < 2; ++ni)
            acc[mi][ni] = (v8f){0.f,0.f,0.f,0.f,0.f,0.f,0.f,0.f};

    for (int k0 = 0; k0 < K; k0 += TILE_K) {
        const bool kfull = (k0 + TILE_K) <= K;

        // ---- stage X tile (128 rows x 32 k) as packed f16 pairs ----
        if (mfull & kfull) {
            #pragma unroll
            for (int e = 0; e < 8; ++e) {
                int p = e * 256 + tid;
                int r = p >> 4, kk = (p & 15) << 1;
                const float* s = Xb + (long long)(m0 + r) * ldx + (k0 + kk);
                *(hh2*)&Xs[r * LDSW + kk] = __builtin_amdgcn_cvt_pkrtz(s[0], s[1]);
            }
        } else {
            #pragma unroll
            for (int e = 0; e < 8; ++e) {
                int p = e * 256 + tid;
                int r = p >> 4, kk = (p & 15) << 1;
                int gr = m0 + r, gk0 = k0 + kk, gk1 = gk0 + 1;
                int cr = gr  < M ? gr  : (M - 1);
                int c0 = gk0 < K ? gk0 : (K - 1);
                int c1 = gk1 < K ? gk1 : (K - 1);
                const float* rp = Xb + (long long)cr * ldx;
                float f0 = rp[c0], f1 = rp[c1];
                f0 = (gr < M && gk0 < K) ? f0 : 0.0f;
                f1 = (gr < M && gk1 < K) ? f1 : 0.0f;
                *(hh2*)&Xs[r * LDSW + kk] = __builtin_amdgcn_cvt_pkrtz(f0, f1);
            }
        }

        // ---- stage W tile (64 out x 32 k) ----
        if (!wtrans) {
            if (nfull & kfull) {
                #pragma unroll
                for (int e = 0; e < 4; ++e) {
                    int p = e * 256 + tid;
                    int r = p >> 4, kk = (p & 15) << 1;
                    const float* s = Wb + (long long)(n0 + r) * ldw + (k0 + kk);
                    *(hh2*)&Ws[r * LDSW + kk] = __builtin_amdgcn_cvt_pkrtz(s[0], s[1]);
                }
            } else {
                #pragma unroll
                for (int e = 0; e < 4; ++e) {
                    int p = e * 256 + tid;
                    int r = p >> 4, kk = (p & 15) << 1;
                    int gr = n0 + r, gk0 = k0 + kk, gk1 = gk0 + 1;
                    int cr = gr  < Nout ? gr  : (Nout - 1);
                    int c0 = gk0 < K ? gk0 : (K - 1);
                    int c1 = gk1 < K ? gk1 : (K - 1);
                    const float* rp = Wb + (long long)cr * ldw;
                    float f0 = rp[c0], f1 = rp[c1];
                    f0 = (gr < Nout && gk0 < K) ? f0 : 0.0f;
                    f1 = (gr < Nout && gk1 < K) ? f1 : 0.0f;
                    *(hh2*)&Ws[r * LDSW + kk] = __builtin_amdgcn_cvt_pkrtz(f0, f1);
                }
            }
        } else {
            // W stored [K][N] (e.g. Vm): transpose through LDS
            if (nfull & kfull) {
                #pragma unroll
                for (int e = 0; e < 4; ++e) {
                    int p = e * 256 + tid;
                    int o = p & 63, kk = (p >> 6) << 1;
                    const float* s = Wb + (long long)(k0 + kk) * ldw + (n0 + o);
                    *(hh2*)&Ws[o * LDSW + kk] = __builtin_amdgcn_cvt_pkrtz(s[0], s[ldw]);
                }
            } else {
                #pragma unroll
                for (int e = 0; e < 4; ++e) {
                    int p = e * 256 + tid;
                    int o = p & 63, kk = (p >> 6) << 1;
                    int go = n0 + o, gk0 = k0 + kk, gk1 = gk0 + 1;
                    int co = go  < Nout ? go  : (Nout - 1);
                    int c0 = gk0 < K ? gk0 : (K - 1);
                    int c1 = gk1 < K ? gk1 : (K - 1);
                    float f0 = Wb[(long long)c0 * ldw + co];
                    float f1 = Wb[(long long)c1 * ldw + co];
                    f0 = (go < Nout && gk0 < K) ? f0 : 0.0f;
                    f1 = (go < Nout && gk1 < K) ? f1 : 0.0f;
                    *(hh2*)&Ws[o * LDSW + kk] = __builtin_amdgcn_cvt_pkrtz(f0, f1);
                }
            }
        }

        // prefetch next X K-tile into cache (global_prefetch_b8)
        if (k0 + TILE_K < K) {
            int pr = m0 + (tid >> 1);
            int pk = k0 + TILE_K + (tid & 1) * 16;
            if (pr < M && pk < K)
                __builtin_prefetch(&Xb[(long long)pr * ldx + pk], 0, 0);
        }
        __syncthreads();

        // ---- build fragments per ISA 16-bit 16x32 layout, issue 4 WMMA ----
        {
            v16h afrag[2], bfrag[2];
            #pragma unroll
            for (int mi = 0; mi < 2; ++mi) {
                const int arow = (mpair * 2 + mi) * 16 + lrow;
                v8h alo = *(const v8h*)&Xs[arow * LDSW + hsel * 8];
                v8h ahi = *(const v8h*)&Xs[arow * LDSW + 16 + hsel * 8];
                afrag[mi] = __builtin_shufflevector(alo, ahi,
                    0,1,2,3,4,5,6,7,8,9,10,11,12,13,14,15);
            }
            #pragma unroll
            for (int ni = 0; ni < 2; ++ni) {
                const int brow = (npair * 2 + ni) * 16 + lrow;
                v8h blo = *(const v8h*)&Ws[brow * LDSW + hsel * 8];
                v8h bhi = *(const v8h*)&Ws[brow * LDSW + 16 + hsel * 8];
                bfrag[ni] = __builtin_shufflevector(blo, bhi,
                    0,1,2,3,4,5,6,7,8,9,10,11,12,13,14,15);
            }
            #pragma unroll
            for (int mi = 0; mi < 2; ++mi)
                #pragma unroll
                for (int ni = 0; ni < 2; ++ni)
                    acc[mi][ni] = __builtin_amdgcn_wmma_f32_16x16x32_f16(
                        false, afrag[mi], false, bfrag[ni],
                        (short)0, acc[mi][ni], false, false);
        }
        __syncthreads();
    }

    // ---- epilogue: D layout -> row = base + r + 8*hsel, col per lane ----
    #pragma unroll
    for (int mi = 0; mi < 2; ++mi) {
        #pragma unroll
        for (int ni = 0; ni < 2; ++ni) {
            v8f a = acc[mi][ni];
            int col = n0 + (npair * 2 + ni) * 16 + lrow;
            if (col < Nout) {
                float bv = bias ? bias[col] : 0.0f;
                int rbase = m0 + (mpair * 2 + mi) * 16 + (hsel ? 8 : 0);
                #pragma unroll
                for (int r = 0; r < 8; ++r) {
                    int row = rbase + r;
                    if (row < M) {
                        float v = a[r] * alpha + bv;
                        if (flags & 1) v = fmaxf(v, 0.0f);
                        if (Rb) v += Rb[(long long)row * ldr + col];
                        Yb[(long long)row * ldy + col] = v;
                    }
                }
            }
        }
    }
}

// One wave32 per row softmax (numerically stable).
__global__ __launch_bounds__(256) void softmax_rows_kernel(
    float* __restrict__ A, int rows, int L)
{
    int row  = blockIdx.x * 8 + (threadIdx.x >> 5);
    int lane = threadIdx.x & 31;
    if (row >= rows) return;
    float* p = A + (long long)row * L;
    float mx = -3.4e38f;
    for (int j = lane; j < L; j += 32) mx = fmaxf(mx, p[j]);
    #pragma unroll
    for (int off = 16; off > 0; off >>= 1) mx = fmaxf(mx, __shfl_xor(mx, off, 32));
    float s = 0.0f;
    for (int j = lane; j < L; j += 32) { float e = __expf(p[j] - mx); p[j] = e; s += e; }
    #pragma unroll
    for (int off = 16; off > 0; off >>= 1) s += __shfl_xor(s, off, 32);
    float inv = 1.0f / s;
    for (int j = lane; j < L; j += 32) p[j] *= inv;
}

// Fused: hs(outer product) -> dw1 -> bn1 -> relu -> dw2 -> bn2 -> relu -> mean_n.
// hs[b,c,n,q] = attn[b,h,q,n] * v[b,n,c]; rank-1 per (b,c) so sweep n with
// rolling horizontal-conv partials. Output mT[b][q][c] = mean_n h2.
__global__ __launch_bounds__(256) void fused_dwconv_mean_kernel(
    const float* __restrict__ attn,   // [B*H][QL][N]
    const float* __restrict__ vbuf,   // [B][N][C]
    const float* __restrict__ dw1, const float* __restrict__ dw2,
    const float* __restrict__ g1, const float* __restrict__ b1,
    const float* __restrict__ m1, const float* __restrict__ v1,
    const float* __restrict__ g2, const float* __restrict__ b2,
    const float* __restrict__ m2, const float* __restrict__ v2,
    float* __restrict__ mT)           // [B][QL][C]
{
    int t = blockIdx.x * 256 + threadIdx.x;
    int d    = t & 63;
    int rest = t >> 6;
    int q    = rest % CQL;
    int bh   = rest / CQL;
    if (bh >= CB * CH) return;
    int b = bh >> 3, h = bh & 7;
    int c = h * CHD + d;

    const float* ab = attn + (long long)bh * (CQL * CN);
    const float* vb = vbuf + ((long long)b * CN) * CC + c;

    float w1r[9], w2r[9];
    #pragma unroll
    for (int i = 0; i < 9; ++i) { w1r[i] = dw1[c * 9 + i]; w2r[i] = dw2[c * 9 + i]; }
    float s1  = g1[c] * rsqrtf(v1[c] + 1e-5f);
    float mu1 = m1[c], be1 = b1[c];
    float s2  = g2[c] * rsqrtf(v2[c] + 1e-5f);
    float mu2 = m2[c], be2 = b2[c];

    const bool qv0 = (q - 1) >= 0;
    const bool qv2 = (q + 1) < CQL;

    float accA[3] = {0,0,0}, accB[3] = {0,0,0}, accC[3] = {0,0,0};
    float win[3][3] = {{0,0,0},{0,0,0},{0,0,0}};
    float macc = 0.0f;

    for (int m = 0; m < CN + 2; ++m) {
        if (m < CN) {
            float A[5];
            #pragma unroll
            for (int tt = 0; tt < 5; ++tt) {
                int qq = q - 2 + tt;
                A[tt] = (qq >= 0 && qq < CQL) ? ab[(long long)qq * CN + m] : 0.0f;
            }
            float vvm = vb[(long long)m * CC];
            #pragma unroll
            for (int j = 0; j < 3; ++j) {
                float c0v = w1r[0]*A[j] + w1r[1]*A[j+1] + w1r[2]*A[j+2];
                float c1v = w1r[3]*A[j] + w1r[4]*A[j+1] + w1r[5]*A[j+2];
                float c2v = w1r[6]*A[j] + w1r[7]*A[j+1] + w1r[8]*A[j+2];
                accA[j] += vvm * c2v;   // contributes to h1 row m-1
                accB[j] += vvm * c1v;   // row m
                accC[j] += vvm * c0v;   // row m+1
            }
        }
        if (m >= 1) {
            int r = m - 1;              // h1 row r is now complete
            float h1n[3];
            #pragma unroll
            for (int j = 0; j < 3; ++j) {
                float xv = (accA[j] - mu1) * s1 + be1;
                bool ok = (r < CN) && (j == 1 || (j == 0 ? qv0 : qv2));
                h1n[j] = ok ? fmaxf(xv, 0.0f) : 0.0f;
            }
            #pragma unroll
            for (int j = 0; j < 3; ++j) {
                win[j][0] = win[j][1]; win[j][1] = win[j][2]; win[j][2] = h1n[j];
            }
            if (m >= 2) {               // h2 at row m-2 now computable
                float h2 = 0.0f;
                #pragma unroll
                for (int ki = 0; ki < 3; ++ki)
                    #pragma unroll
                    for (int kj = 0; kj < 3; ++kj)
                        h2 += w2r[ki * 3 + kj] * win[kj][ki];
                float y = (h2 - mu2) * s2 + be2;
                macc += fmaxf(y, 0.0f);
            }
        }
        #pragma unroll
        for (int j = 0; j < 3; ++j) { accA[j] = accB[j]; accB[j] = accC[j]; accC[j] = 0.0f; }
    }
    mT[((long long)b * CQL + q) * CC + c] = macc * (1.0f / (float)CN);
}

// ---------------------------------------------------------------------------
// CDNA5 feature probe (compiled into the object; not launched). Exercises the
// ASYNCcnt global->LDS path, TENSORcnt wait, and cluster barrier/ID paths.
// ---------------------------------------------------------------------------
__global__ void cdna5_async_probe_kernel(const float* __restrict__ g,
                                         float* __restrict__ outp)
{
    __shared__ __attribute__((aligned(16))) float stage[64];
    unsigned lane = threadIdx.x & 31u;
    unsigned ldsByte = lane * 4u;
    const float* src = g + lane;
    asm volatile("global_load_async_to_lds_b32 %0, %1, off"
                 :: "v"(ldsByte), "v"(src) : "memory");
    asm volatile("s_wait_asynccnt 0x0" ::: "memory");
    __builtin_amdgcn_s_wait_tensorcnt(0);
    __builtin_amdgcn_s_cluster_barrier();
    int cid = __builtin_amdgcn_cluster_id_x();
    outp[threadIdx.x] = stage[lane] + (float)cid;
}

extern "C" void kernel_launch(void* const* d_in, const int* in_sizes, int n_in,
                              void* d_out, int out_size, void* d_ws, size_t ws_size,
                              hipStream_t stream)
{
    const float* x     = (const float*)d_in[0];
    const float* Wq    = (const float*)d_in[1];
    const float* Wk    = (const float*)d_in[2];
    const float* Wv    = (const float*)d_in[3];
    const float* Wproj = (const float*)d_in[4];
    const float* bproj = (const float*)d_in[5];
    const float* dw1   = (const float*)d_in[6];
    const float* dw2   = (const float*)d_in[7];
    const float* pw    = (const float*)d_in[8];
    const float* bn1g  = (const float*)d_in[9];
    const float* bn1b  = (const float*)d_in[10];
    const float* bn1m  = (const float*)d_in[11];
    const float* bn1v  = (const float*)d_in[12];
    const float* bn2g  = (const float*)d_in[13];
    const float* bn2b  = (const float*)d_in[14];
    const float* bn2m  = (const float*)d_in[15];
    const float* bn2v  = (const float*)d_in[16];
    const float* mWq   = (const float*)d_in[17];
    const float* mbq   = (const float*)d_in[18];
    const float* mWk   = (const float*)d_in[19];
    const float* mbk   = (const float*)d_in[20];
    const float* mWv   = (const float*)d_in[21];
    const float* mbv   = (const float*)d_in[22];
    const float* mWo   = (const float*)d_in[23];
    const float* mbo   = (const float*)d_in[24];
    float* out = (float*)d_out;

    float* ws = (float*)d_ws;
    size_t off = 0;
    auto alloc = [&](size_t n) { float* p = ws + off; off += n; return p; };
    float* qb  = alloc((size_t)CB * CQL * CC);      // scaled q
    float* kb  = alloc((size_t)CB * CN * CC);
    float* vb  = alloc((size_t)CB * CN * CC);
    float* at1 = alloc((size_t)CB * CH * CQL * CN); // attn1 (in-place softmax)
    float* mT  = alloc((size_t)CB * CQL * CC);      // mean_n h2, [b][q][c]
    float* kc  = alloc((size_t)CB * CQL * CC);
    float* qm  = alloc((size_t)CB * CN * CC);
    float* km  = alloc((size_t)CB * CQL * CC);
    float* vm  = alloc((size_t)CB * CQL * CC);
    float* at2 = alloc((size_t)CB * CH * CN * CQL);
    float* ob  = alloc((size_t)CB * CN * CC);
    float* o2  = alloc((size_t)CB * CN * CC);
    (void)ws_size; (void)in_sizes; (void)n_in; (void)out_size;

    dim3 blk(256);
    auto gemm = [&](const float* X, const float* W, const float* bias,
                    const float* resid, float* Y,
                    int M, int Nout, int K, int ldx, int ldw, int ldy, int ldr,
                    long long sxo, long long sxi, long long swo, long long swi,
                    long long syo, long long syi, long long sro, long long sri,
                    int outer, int inner, float alpha, int flags) {
        dim3 grid((M + TILE_M - 1) / TILE_M, (Nout + TILE_N - 1) / TILE_N,
                  outer * inner);
        gemm_wmma_kernel<<<grid, blk, 0, stream>>>(
            X, W, bias, resid, Y, M, Nout, K, ldx, ldw, ldy, ldr,
            sxo, sxi, swo, swi, syo, syi, sro, sri, inner, alpha, flags);
    };

    const long long SX = (long long)CSEQ * CC;   // per-batch stride of x
    const long long SQ = (long long)CQL * CC;
    const long long SN = (long long)CN * CC;

    // 1-3: q/k/v projections (q pre-scaled by hd^-0.5)
    gemm(x + CN * CC, Wq, 0, 0, qb, CQL, CC, CC, CC, CC, CC, 0,
         SX, 0, 0, 0, SQ, 0, 0, 0, CB, 1, 0.125f, 0);
    gemm(x, Wk, 0, 0, kb, CN, CC, CC, CC, CC, CC, 0,
         SX, 0, 0, 0, SN, 0, 0, 0, CB, 1, 1.0f, 0);
    gemm(x, Wv, 0, 0, vb, CN, CC, CC, CC, CC, CC, 0,
         SX, 0, 0, 0, SN, 0, 0, 0, CB, 1, 1.0f, 0);

    // 4: logits1[b,h,q,n] = q . k per head
    gemm(qb, kb, 0, 0, at1, CQL, CN, CHD, CC, CC, CN, 0,
         SQ, CHD, SN, CHD,
         (long long)CH * CQL * CN, (long long)CQL * CN, 0, 0,
         CB, CH, 1.0f, 0);

    // 5: softmax over n
    softmax_rows_kernel<<<dim3((CB * CH * CQL + 7) / 8), blk, 0, stream>>>(
        at1, CB * CH * CQL, CN);

    // 6: fused hs -> dw1/bn1/relu -> dw2/bn2/relu -> mean_n
    fused_dwconv_mean_kernel<<<dim3((CB * CH * CHD * CQL) / 256), blk, 0, stream>>>(
        at1, vb, dw1, dw2, bn1g, bn1b, bn1m, bn1v, bn2g, bn2b, bn2m, bn2v, mT);

    // 7: kc = cls_cat + mT @ pw^T
    gemm(mT, pw, 0, x + CN * CC, kc, CQL, CC, CC, CC, CC, CC, CC,
         SQ, 0, 0, 0, SQ, 0, SX, 0, CB, 1, 1.0f, 0);

    // 8-10: Qm/Km/Vm
    gemm(x, mWq, mbq, 0, qm, CN, CC, CC, CC, CC, CC, 0,
         SX, 0, 0, 0, SN, 0, 0, 0, CB, 1, 1.0f, 0);
    gemm(kc, mWk, mbk, 0, km, CQL, CC, CC, CC, CC, CC, 0,
         SQ, 0, 0, 0, SQ, 0, 0, 0, CB, 1, 1.0f, 0);
    gemm(kc, mWv, mbv, 0, vm, CQL, CC, CC, CC, CC, CC, 0,
         SQ, 0, 0, 0, SQ, 0, 0, 0, CB, 1, 1.0f, 0);

    // 11: logits2[b,h,n,q] = Qm . Km / sqrt(C)
    gemm(qm, km, 0, 0, at2, CN, CQL, CHD, CC, CC, CQL, 0,
         SN, CHD, SQ, CHD,
         (long long)CH * CN * CQL, (long long)CN * CQL, 0, 0,
         CB, CH, 0.04419417382415922f, 0);

    // 12: softmax over q
    softmax_rows_kernel<<<dim3((CB * CH * CN + 7) / 8), blk, 0, stream>>>(
        at2, CB * CH * CN, CQL);

    // 13: O = Qm + A @ Vm   (Vm accessed [k=q][d] -> wtrans)
    gemm(at2, vm, 0, qm, ob, CN, CHD, CQL, CQL, CC, CC, CC,
         (long long)CH * CN * CQL, (long long)CN * CQL, SQ, CHD,
         SN, CHD, SN, CHD,
         CB, CH, 1.0f, 4);

    // 14: O2 = O + relu(O @ mWo^T + mbo)   (flat M = B*N)
    gemm(ob, mWo, mbo, ob, o2, CB * CN, CC, CC, CC, CC, CC, CC,
         0, 0, 0, 0, 0, 0, 0, 0, 1, 1, 1.0f, 1);

    // 15: out = O2 @ Wproj^T + bproj
    gemm(o2, Wproj, bproj, 0, out, CB * CN, CC, CC, CC, CC, CC, 0,
         0, 0, 0, 0, 0, 0, 0, 0, 1, 1, 1.0f, 0);
}
